// CausalContagionPredictor_4329327035072
// MI455X (gfx1250) — matrix operand order repaired and checked by reading, not compiled, not run on hardware
//
#include <hip/hip_runtime.h>
#include <hip/hip_bf16.h>
#include <math.h>

typedef __attribute__((ext_vector_type(16))) _Float16 v16h;
typedef __attribute__((ext_vector_type(8)))  float    v8f;
typedef __attribute__((ext_vector_type(4)))  unsigned int u32x4;
typedef __attribute__((ext_vector_type(4)))  int      i32x4;
typedef __attribute__((ext_vector_type(8)))  int      i32x8;

#define N_NODES   512
#define HIDDEN    64
#define H2        32
#define MAX_STEPS 10

// ---- workspace byte offsets ----
#define OFF_SRCPROJ 0u            // 512*64 f32 = 131072 B
#define OFF_TGTPROJ 131072u       // 512*64 f32
#define OFF_FD      262144u       // 512*512 f32 = 1 MB
#define OFF_P       1310720u      // 512 f32
#define OFF_ARR     1312768u      // 512 f32
#define OFF_CAND    1314816u      // 512 f32
#define OFF_W2PK    1316864u      // 4 ops * 32 lanes * 16 halves = 4096 B (32B aligned)

// ds_swizzle group-of-32 XOR butterflies (and_mask=0x1f, xor in bits [14:10])
#define SWZ_XOR(x, imm) \
    __int_as_float(__builtin_amdgcn_ds_swizzle(__float_as_int(x), (imm)))

// ---------------- init: p0 / arr0 / cand ----------------
__global__ void ccp_init_kernel(const int* __restrict__ shock,
                                float* __restrict__ p, float* __restrict__ arr,
                                float* __restrict__ cand) {
    int j = threadIdx.x;              // 512 threads, 1 block
    p[j] = 0.0f;
    arr[j] = __builtin_inff();
    cand[j] = 0.0f;
    __syncthreads();
    if (j < 4) {
        int s = shock[j];
        p[s] = 1.0f;
        arr[s] = 0.0f;
    }
}

// ---------------- precompute src/tgt first-layer projections ----------------
__global__ void ccp_proj_kernel(const float* __restrict__ nf,
                                const float* __restrict__ W1,
                                float* __restrict__ srcproj,
                                float* __restrict__ tgtproj) {
    int gid = blockIdx.x * blockDim.x + threadIdx.x;   // 65536 threads
    int half = gid >> 15;                              // 0 = src, 1 = tgt
    int o    = gid & 32767;
    int i = o >> 6, d = o & 63;
    const float* nrow = nf + i * HIDDEN;
    const float* wcol = W1 + half * HIDDEN * HIDDEN + d; // rows [0,64) or [64,128)
    float acc = 0.0f;
#pragma unroll
    for (int k = 0; k < HIDDEN; ++k)
        acc = fmaf(nrow[k], wcol[k * HIDDEN], acc);
    if (half == 0) srcproj[o] = acc; else tgtproj[o] = acc;
}

// ------- precompute fd[i][j]; pack W2 into wave32 WMMA-B operand layout -------
// B order used here (contiguous k, same as A): lane L holds N = nt*16 + (L&15), K = c*32 + (L>>4)*16 + h.
// Packed order: w2pk[((c*2+nt)*32 + lane)*16 + h]  -> per-lane contiguous 32B v16h.
__global__ void ccp_fd_w2_kernel(const float* __restrict__ nf,
                                 const float* __restrict__ W2,
                                 float* __restrict__ fd,
                                 _Float16* __restrict__ w2pk) {
    int gid = blockIdx.x * blockDim.x + threadIdx.x;   // 262144 threads
    int i = gid >> 9, j = gid & 511;
    fd[gid] = fabsf(nf[i * HIDDEN] - nf[j * HIDDEN]);
    if (gid < 128) {
        const int op = gid >> 5;        // 0..3 = c*2+nt
        const int ln = gid & 31;
        const int c  = op >> 1, nt = op & 1;
        _Float16* dst = w2pk + (op * 32 + ln) * 16;
#pragma unroll
        for (int h = 0; h < 16; ++h)
            dst[h] = (_Float16)W2[(c * 32 + (ln >> 4) * 16 + h) * H2
                                  + nt * 16 + (ln & 15)];
    }
}

// ---------------- per-step edge MLP (WMMA + TDM) + scatter-max ----------------
// block = 256 threads = 8 waves; block covers 1 source i x 128 targets j
__global__ __launch_bounds__(256)
void ccp_edge_kernel(const float* __restrict__ cg,
                     const float* __restrict__ W1,
                     const float* __restrict__ b1,
                     const float* __restrict__ b2,
                     const float* __restrict__ W3,
                     const float* __restrict__ b3,
                     const float* __restrict__ srcproj,
                     const float* __restrict__ tgtproj,
                     const float* __restrict__ fd,
                     const _Float16* __restrict__ w2pk,
                     const float* __restrict__ p,
                     float* __restrict__ cand,
                     float stepfrac) {
    const int i      = blockIdx.x >> 2;
    const int jblock = (blockIdx.x & 3) * 128;
    const int tid    = threadIdx.x;

    const float p_i = p[i];
    if (p_i <= 0.0f) return;            // uniform over block: inactive source row

    __shared__ __align__(16) float base_s[HIDDEN];
    __shared__ __align__(16) float w1r0_s[HIDDEN];
    __shared__ __align__(16) float w1r3_s[HIDDEN];
    __shared__ __align__(16) float cg_s[128];
    __shared__ __align__(16) float fd_s[128];
    __shared__ __align__(16) float traw_s[128 * HIDDEN];  // raw tgtproj tile, 32 KB

    // ---- TDM: DMA the 128x64 f32 tgtproj tile into LDS (wave 0 issues) ----
    if (tid < 32) {
        const unsigned long long gaddr =
            (unsigned long long)(const void*)(tgtproj + (unsigned)jblock * HIDDEN);
        const unsigned int ldsa = (unsigned int)(unsigned long long)(const void*)traw_s;

        u32x4 g0;
        g0[0] = 1u;                                         // count=1, user D#
        g0[1] = ldsa;                                       // lds_addr (bytes)
        g0[2] = (unsigned int)(gaddr & 0xFFFFFFFFu);        // global_addr[31:0]
        g0[3] = (unsigned int)(gaddr >> 32) | (2u << 30);   // addr[56:32] | type=2

        i32x8 g1;
        g1[0] = 0x20000;               // data_size=2 (4B), no multicast/pad/iterate
        g1[1] = (int)(64u << 16);      // tensor_dim0 = 64   ([79:48] low 16)
        g1[2] = (int)(128u << 16);     // tensor_dim1 = 128  ([111:80] low 16)
        g1[3] = (int)(64u << 16);      // tile_dim0 = 64     ([127:112])
        g1[4] = 128;                   // tile_dim1 = 128, tile_dim2 = 0
        g1[5] = 64;                    // tensor_dim0_stride = 64 (low 32)
        g1[6] = 0;                     // stride hi / tensor_dim1_stride lo
        g1[7] = 0;

        i32x4 g2 = {0, 0, 0, 0};
        i32x4 g3 = {0, 0, 0, 0};
#if __clang_major__ >= 23
        i32x8 g4 = {0, 0, 0, 0, 0, 0, 0, 0};
        __builtin_amdgcn_tensor_load_to_lds(g0, g1, g2, g3, g4, 0);
#else
        __builtin_amdgcn_tensor_load_to_lds(g0, g1, g2, g3, 0);
#endif
    }

    // ---- phase 1 (overlaps the DMA): per-block vectors ----
    if (tid < HIDDEN) {
        float sp = srcproj[i * HIDDEN + tid];
        base_s[tid] = sp + p_i * W1[129 * HIDDEN + tid]
                         + stepfrac * W1[130 * HIDDEN + tid] + b1[tid];
        w1r0_s[tid] = W1[128 * HIDDEN + tid];
        w1r3_s[tid] = W1[131 * HIDDEN + tid];
    }
    if (tid < 128) {
        cg_s[tid] = cg[i * N_NODES + jblock + tid];
        fd_s[tid] = fd[i * N_NODES + jblock + tid];
    }

    __builtin_amdgcn_s_wait_tensorcnt(0);   // wave 0 drains its DMA; no-op elsewhere
    __syncthreads();

    const int wave = tid >> 5;          // 0..7 -> j tile
    const int lane = tid & 31;
    const int g    = lane >> 4;         // K-half select for A, K-row group for B
    const int m    = lane & 15;         // A row (edge within tile) for this lane
    const int j0   = jblock + wave * 16;
    const int jl   = wave * 16 + m;     // LDS-local target index for A build

    // ---- B operands: pre-packed, one contiguous v16h per lane per operand ----
    const v16h* w2p = (const v16h*)w2pk;
    const v16h b00 = w2p[0 * 32 + lane];   // c=0, ntile=0
    const v16h b01 = w2p[1 * 32 + lane];   // c=0, ntile=1
    const v16h b10 = w2p[2 * 32 + lane];   // c=1, ntile=0
    const v16h b11 = w2p[3 * 32 + lane];   // c=1, ntile=1

    // ---- build A = relu(h1) tile (16 edges x 64): float2 per A-VGPR ----
    const float cgv = cg_s[jl];
    const float fdv = fd_s[jl];
    const float* trw = traw_s + jl * HIDDEN;

    v16h a[2];
#pragma unroll
    for (int c = 0; c < 2; ++c) {
#pragma unroll
        for (int v = 0; v < 8; ++v) {
            const int kb = c * 32 + g * 16 + v * 2;
            const float2 t2 = *(const float2*)&trw[kb];
            const float2 bs = *(const float2*)&base_s[kb];
            const float2 r0 = *(const float2*)&w1r0_s[kb];
            const float2 r3 = *(const float2*)&w1r3_s[kb];
            const float hx = t2.x + bs.x + cgv * r0.x + fdv * r3.x;
            const float hy = t2.y + bs.y + cgv * r0.y + fdv * r3.y;
            a[c][2 * v]     = (_Float16)fmaxf(hx, 0.0f);
            a[c][2 * v + 1] = (_Float16)fmaxf(hy, 0.0f);
        }
    }

    // ---- h2 = relu(h1) @ W2 : 4x V_WMMA_F32_16X16X32_F16, f32 accum ----
    v8f c0 = {}, c1 = {};
    c0 = __builtin_amdgcn_wmma_f32_16x16x32_f16(false, a[0], false, b00, (short)0, c0, false, false);
    c0 = __builtin_amdgcn_wmma_f32_16x16x32_f16(false, a[1], false, b10, (short)0, c0, false, false);
    c1 = __builtin_amdgcn_wmma_f32_16x16x32_f16(false, a[0], false, b01, (short)0, c1, false, false);
    c1 = __builtin_amdgcn_wmma_f32_16x16x32_f16(false, a[1], false, b11, (short)0, c1, false, false);

    // ---- epilogue: +b2, relu, dot W3, ds_swizzle butterfly, sigmoid ----
    const int n0 = m, n1 = 16 + m;
    const float b2v0 = b2[n0], b2v1 = b2[n1];
    const float w3v0 = W3[n0], w3v1 = W3[n1];
    const float b3v  = b3[0];

    float tvals[8];
#pragma unroll
    for (int r = 0; r < 8; ++r) {
        float s = fmaxf(c0[r] + b2v0, 0.0f) * w3v0
                + fmaxf(c1[r] + b2v1, 0.0f) * w3v1;
        s += SWZ_XOR(s, 0x041f);   // SWAPX1
        s += SWZ_XOR(s, 0x081f);   // SWAPX2
        s += SWZ_XOR(s, 0x101f);   // SWAPX4
        s += SWZ_XOR(s, 0x201f);   // SWAPX8 (stays within 16-lane half)
        tvals[r] = 1.0f / (1.0f + __expf(-(s + b3v)));
    }

    if (m == 0) {   // lanes 0 and 16: one emitter per C row-group
#pragma unroll
        for (int r = 0; r < 8; ++r) {
            const int mm = g * 8 + r;            // edge within tile
            const float cge = cg_s[wave * 16 + mm];
            if (cge > 0.0f) {
                const float np = p_i * tvals[r] * cge;
                if (np > 0.0f)
                    atomicMax((unsigned int*)&cand[j0 + mm], __float_as_uint(np));
            }
        }
    }
}

// ---------------- per-step relaxation of p / arr, reset cand ----------------
__global__ void ccp_update_kernel(float* __restrict__ p, float* __restrict__ arr,
                                  float* __restrict__ cand, float step_p1) {
    int j = blockIdx.x * blockDim.x + threadIdx.x;
    if (j < N_NODES) {
        float c = cand[j], pv = p[j];
        if (c > pv) {
            p[j] = c;
            arr[j] = fminf(arr[j], step_p1);
        }
        cand[j] = 0.0f;
    }
}

// ---------------- emit (p_final, arr_final) ----------------
__global__ void ccp_final_kernel(const float* __restrict__ p,
                                 const float* __restrict__ arr,
                                 float* __restrict__ out) {
    int j = threadIdx.x;              // 512 threads, 1 block
    out[j] = p[j];
    out[N_NODES + j] = arr[j];
}

extern "C" void kernel_launch(void* const* d_in, const int* in_sizes, int n_in,
                              void* d_out, int out_size, void* d_ws, size_t ws_size,
                              hipStream_t stream) {
    const float* cg    = (const float*)d_in[0];
    const float* nf    = (const float*)d_in[1];
    const int*   shock = (const int*)  d_in[2];
    const float* W1    = (const float*)d_in[3];
    const float* b1    = (const float*)d_in[4];
    const float* W2    = (const float*)d_in[5];
    const float* b2    = (const float*)d_in[6];
    const float* W3    = (const float*)d_in[7];
    const float* b3    = (const float*)d_in[8];
    float* out = (float*)d_out;

    char* ws = (char*)d_ws;
    float*     srcproj = (float*)(ws + OFF_SRCPROJ);
    float*     tgtproj = (float*)(ws + OFF_TGTPROJ);
    float*     fd      = (float*)(ws + OFF_FD);
    float*     p       = (float*)(ws + OFF_P);
    float*     arr     = (float*)(ws + OFF_ARR);
    float*     cand    = (float*)(ws + OFF_CAND);
    _Float16*  w2pk    = (_Float16*)(ws + OFF_W2PK);

    ccp_init_kernel<<<1, N_NODES, 0, stream>>>(shock, p, arr, cand);
    ccp_proj_kernel<<<256, 256, 0, stream>>>(nf, W1, srcproj, tgtproj);
    ccp_fd_w2_kernel<<<1024, 256, 0, stream>>>(nf, W2, fd, w2pk);

    for (int s = 0; s < MAX_STEPS; ++s) {
        const float stepfrac = (float)s / (float)MAX_STEPS;
        ccp_edge_kernel<<<N_NODES * 4, 256, 0, stream>>>(
            cg, W1, b1, b2, W3, b3, srcproj, tgtproj, fd, w2pk, p, cand, stepfrac);
        ccp_update_kernel<<<2, 256, 0, stream>>>(p, arr, cand, (float)(s + 1));
    }

    ccp_final_kernel<<<1, N_NODES, 0, stream>>>(p, arr, out);
}
